// BahdanauAttention_61795989455450
// MI455X (gfx1250) — matrix operand order repaired and checked by reading, not compiled, not run on hardware
//
#include <hip/hip_runtime.h>

#define B_  32
#define S_  2048
#define HD_ 1024
#define HE_ 1024

#define ROWS_WG    64          // rows of flattened (b,s) per workgroup (4 waves x 16)
#define LDS_STRIDE 1032        // 1024 + 8 pad elems -> 2064B row stride, bank-conflict-free

typedef __attribute__((ext_vector_type(16))) __bf16 v16bf;
typedef __attribute__((ext_vector_type(8)))  float  v8f;

union BFrag {
    v16bf v;
    unsigned short s[16];
    uint4 q[2];
};

__device__ __forceinline__ unsigned short f2b(float f) {
    unsigned u = __float_as_uint(f);
    u += 0x7FFFu + ((u >> 16) & 1u);     // round-to-nearest-even
    return (unsigned short)(u >> 16);
}

__device__ __forceinline__ float fast_tanh(float x) {
#if __has_builtin(__builtin_amdgcn_tanhf)
    return __builtin_amdgcn_tanhf(x);    // v_tanh_f32 (CDNA5 TRANS op)
#else
    return tanhf(x);
#endif
}

// ---------------------------------------------------------------------------
// Stage 0a: W_key f32 -> bf16 (2MB, stays L2-resident for the GEMM stage)
// ---------------------------------------------------------------------------
__global__ void __launch_bounds__(256) k_convert_wkey(const float* __restrict__ wk,
                                                      unsigned short* __restrict__ wkb) {
    int i = (blockIdx.x * 256 + threadIdx.x) * 4;   // 1024 blocks cover 1M elems
    float4 f = *(const float4*)(wk + i);
    uint2 o;
    o.x = (unsigned)f2b(f.x) | ((unsigned)f2b(f.y) << 16);
    o.y = (unsigned)f2b(f.z) | ((unsigned)f2b(f.w) << 16);
    *(uint2*)(wkb + i) = o;
}

// ---------------------------------------------------------------------------
// Stage 0b: q[b,d] = sum_e dh[b,e] * Wq[d,e]   (32x1024, 67 MFLOP - tiny)
// ---------------------------------------------------------------------------
__global__ void __launch_bounds__(256) k_query(const float* __restrict__ dh,
                                               const float* __restrict__ Wq,
                                               float* __restrict__ qbuf) {
    __shared__ float sdh[HD_];
    int b   = blockIdx.y;
    int tid = threadIdx.x;
    ((float4*)sdh)[tid] = ((const float4*)(dh + (size_t)b * HD_))[tid];
    __syncthreads();
    int d = blockIdx.x * 256 + tid;
    const float4* wr = (const float4*)(Wq + (size_t)d * HD_);
    const float4* xr = (const float4*)sdh;
    float acc = 0.f;
    #pragma unroll 4
    for (int i = 0; i < HD_ / 4; ++i) {
        float4 w = wr[i], x = xr[i];
        acc += w.x * x.x + w.y * x.y + w.z * x.z + w.w * x.w;
    }
    qbuf[(size_t)b * HD_ + d] = acc;
}

// ---------------------------------------------------------------------------
// Stage 1: fused  scores[b,s] = v . tanh(q[b,:] + enc[b,s,:] @ Wk^T)
// WMMA bf16 16x16x32.  1024 WGs x 4 waves; wave owns 16 (b,s) rows.
// Encoder tile (64 rows x 1024) converted to bf16 ONCE into LDS, then
// A frags come from ds_load_b128; B frags stream from L2-resident bf16 W_key.
// A frag (16x32, 16-bit): lanes 0-15 row=lane K=0..7,16..23 ; lanes 16-31 K=8..15,24..31
// B frag (32x16, 16-bit): lanes 0-15 col=lane K=0..7,16..23 ; lanes 16-31 K=8..15,24..31
// ---------------------------------------------------------------------------
__global__ void __launch_bounds__(128) k_scores(const float* __restrict__ enc,
                                                const unsigned short* __restrict__ wkb,
                                                const float* __restrict__ qbuf,
                                                const float* __restrict__ ven,
                                                float* __restrict__ scores) {
    extern __shared__ unsigned short sA[];      // [ROWS_WG][LDS_STRIDE] bf16, 129 KB

    const int tid    = threadIdx.x;
    const int lane   = tid & 31;
    const int wid    = tid >> 5;
    const int laneM  = lane & 15;
    const int hi     = lane >> 4;               // which K-half this lane holds
    const int wgRow0 = blockIdx.x * ROWS_WG;
    const int m0     = wgRow0 + wid * 16;       // wave's row base (b*S + s)
    const int b      = m0 >> 11;                // S_ == 2048

    // ---- fill: 64 rows x 1024 f32 -> bf16 LDS (coalesced float4 stream) ----
    const float* encBase = enc + (size_t)wgRow0 * HE_;
    #pragma unroll 4
    for (int it = 0; it < (ROWS_WG * HE_ / 4) / 128; ++it) {
        int f4 = it * 128 + tid;
        int r  = f4 >> 8;                       // 256 float4 per row
        int c  = (f4 & 255) * 4;
        float4 f = *(const float4*)(encBase + (size_t)r * HE_ + c);
        uint2 o;
        o.x = (unsigned)f2b(f.x) | ((unsigned)f2b(f.y) << 16);
        o.y = (unsigned)f2b(f.z) | ((unsigned)f2b(f.w) << 16);
        *(uint2*)(sA + r * LDS_STRIDE + c) = o;
    }
    __syncthreads();

    const unsigned short* aRow = sA + (wid * 16 + laneM) * LDS_STRIDE + hi * 8;

    float acc[8] = {0.f, 0.f, 0.f, 0.f, 0.f, 0.f, 0.f, 0.f};
    const v8f vzero = {0.f, 0.f, 0.f, 0.f, 0.f, 0.f, 0.f, 0.f};

    for (int n0 = 0; n0 < HD_; n0 += 128) {
        v8f c[8];
        #pragma unroll
        for (int j = 0; j < 8; ++j) c[j] = vzero;

        #pragma unroll 2
        for (int k0 = 0; k0 < HE_; k0 += 32) {
            // ---- A fragment: two ds_load_b128 from staged bf16 tile ----
            BFrag a;
            a.q[0] = *(const uint4*)(aRow + k0);        // K = ka .. ka+7
            a.q[1] = *(const uint4*)(aRow + k0 + 16);   // K = ka+16 .. ka+23

            // ---- 8 column tiles: B frags straight from L2-resident bf16 W_key ----
            #pragma unroll
            for (int j = 0; j < 8; ++j) {
                int n = n0 + j * 16 + laneM;
                const uint4* bp = (const uint4*)(wkb + (size_t)n * HE_ + k0 + hi * 8);
                BFrag bb;
                bb.q[0] = bp[0];
                bb.q[1] = bp[2];
                c[j] = __builtin_amdgcn_wmma_f32_16x16x32_bf16(
                    false, a.v, false, bb.v, (short)0, c[j], false, false);
            }
        }

        // ---- epilogue for this d-chunk: + q[d], tanh, * v[d], accumulate ----
        #pragma unroll
        for (int j = 0; j < 8; ++j) {
            int n = n0 + j * 16 + laneM;
            float qv = qbuf[(size_t)b * HD_ + n];
            float vv = ven[n];
            #pragma unroll
            for (int i = 0; i < 8; ++i)
                acc[i] += vv * fast_tanh(c[j][i] + qv);
        }
    }

    // C layout: VGPR i holds row m0+i (lanes 0-15) and row m0+8+i (lanes 16-31).
    #pragma unroll
    for (int i = 0; i < 8; ++i) {
        float v = acc[i];
        v += __shfl_xor(v, 1, 32);
        v += __shfl_xor(v, 2, 32);
        v += __shfl_xor(v, 4, 32);
        v += __shfl_xor(v, 8, 32);
        if (laneM == 0)
            scores[m0 + i + (hi << 3)] = v;
    }
}

// ---------------------------------------------------------------------------
// Stage 2: masked softmax over S per batch; also zero-init context row
// ---------------------------------------------------------------------------
__global__ void __launch_bounds__(256) k_softmax(const float* __restrict__ scores,
                                                 const int* __restrict__ mask,
                                                 float* __restrict__ attn,
                                                 float* __restrict__ ctx) {
    __shared__ float red[256];
    int b = blockIdx.x, tid = threadIdx.x;
    const float* sr = scores + (size_t)b * S_;
    const int*   mr = mask   + (size_t)b * S_;

    float v[8], mx = -3.4e38f;
    #pragma unroll
    for (int j = 0; j < 8; ++j) {
        int idx = tid + j * 256;
        float s = sr[idx];
        v[j] = mr[idx] ? s : -1e9f;
        mx = fmaxf(mx, v[j]);
    }
    red[tid] = mx; __syncthreads();
    for (int o = 128; o > 0; o >>= 1) {
        if (tid < o) red[tid] = fmaxf(red[tid], red[tid + o]);
        __syncthreads();
    }
    mx = red[0]; __syncthreads();

    float sum = 0.f;
    #pragma unroll
    for (int j = 0; j < 8; ++j) { v[j] = __expf(v[j] - mx); sum += v[j]; }
    red[tid] = sum; __syncthreads();
    for (int o = 128; o > 0; o >>= 1) {
        if (tid < o) red[tid] += red[tid + o];
        __syncthreads();
    }
    float inv = 1.f / red[0];
    #pragma unroll
    for (int j = 0; j < 8; ++j)
        attn[(size_t)b * S_ + tid + j * 256] = v[j] * inv;

    // zero context row for this batch (d_out is poisoned; atomics accumulate later)
    float4 z = {0.f, 0.f, 0.f, 0.f};
    ((float4*)(ctx + (size_t)b * HD_))[tid] = z;
}

// ---------------------------------------------------------------------------
// Stage 3: context[b,e] = sum_s attn[b,s] * enc[b,s,e]
// grid (e-chunk 8, s-chunk 8, b 32) = 2048 WGs to saturate HBM
// ---------------------------------------------------------------------------
__global__ void __launch_bounds__(128) k_context(const float* __restrict__ enc,
                                                 const float* __restrict__ attn,
                                                 float* __restrict__ ctx) {
    __shared__ float w[256];
    int ec = blockIdx.x, sc = blockIdx.y, b = blockIdx.z;
    int tid = threadIdx.x;
    const float* ar = attn + (size_t)b * S_ + sc * 256;
    w[tid]       = ar[tid];
    w[tid + 128] = ar[tid + 128];
    __syncthreads();

    int e = ec * 128 + tid;
    const float* ep = enc + ((size_t)b * S_ + (size_t)sc * 256) * HE_ + e;
    float acc = 0.f;
    #pragma unroll 4
    for (int s = 0; s < 256; ++s)
        acc += w[s] * ep[(size_t)s * HE_];

    atomicAdd(ctx + (size_t)b * HD_ + e, acc);   // global_atomic_add_f32
}

// ---------------------------------------------------------------------------
extern "C" void kernel_launch(void* const* d_in, const int* in_sizes, int n_in,
                              void* d_out, int out_size, void* d_ws, size_t ws_size,
                              hipStream_t stream) {
    const float* dh   = (const float*)d_in[0];   // [32,1024]
    const float* enc  = (const float*)d_in[1];   // [32,2048,1024]
    const int*   mask = (const int*)d_in[2];     // [32,2048]
    const float* Wq   = (const float*)d_in[3];   // [1024,1024]
    const float* Wk   = (const float*)d_in[4];   // [1024,1024]
    const float* Wv   = (const float*)d_in[5];   // [1,1024]

    float* out  = (float*)d_out;
    float* ctx  = out;                 // [32,1024]
    float* attn = out + B_ * HD_;      // [32,2048]

    float*          qbuf   = (float*)d_ws;                       // 128 KB
    float*          scores = qbuf + B_ * HD_;                    // 256 KB
    unsigned short* wkb    = (unsigned short*)(scores + B_ * S_);// 2 MB bf16 W_key

    k_convert_wkey<<<(HD_ * HE_) / (256 * 4), 256, 0, stream>>>(Wk, wkb);
    k_query<<<dim3(HD_ / 256, B_), 256, 0, stream>>>(dh, Wq, qbuf);

    size_t ldsBytes = (size_t)ROWS_WG * LDS_STRIDE * sizeof(unsigned short); // 129 KB
    k_scores<<<(B_ * S_) / ROWS_WG, 128, ldsBytes, stream>>>(enc, wkb, qbuf, Wv, scores);

    k_softmax<<<B_, 256, 0, stream>>>(scores, mask, attn, ctx);
    k_context<<<dim3(8, 8, B_), 128, 0, stream>>>(enc, attn, ctx);
}
